// TemporalFusionDecoder_61357902790848
// MI455X (gfx1250) — matrix operand order repaired and checked by reading, not compiled, not run on hardware
//
#include <hip/hip_runtime.h>
#include <hip/hip_bf16.h>
#include <math.h>

#define DEVI __device__ __forceinline__

#if defined(__HIP_DEVICE_COMPILE__) && __has_builtin(__builtin_amdgcn_global_load_async_to_lds_b128)
#define USE_ASYNC_LDS 1
#else
#define USE_ASYNC_LDS 0
#endif

namespace {

constexpr int Bq  = 64;
constexpr int Tt  = 1024;
constexpr int Hh  = 256;
constexpr int NHh = 4;
constexpr int DHh = 64;
constexpr int ELl = 768;
constexpr int TDd = Tt - ELl;     // 256
constexpr int Rr  = Bq * Tt;      // 65536
constexpr int RDd = Bq * TDd;     // 16384
constexpr float EPSc = 1e-3f;

typedef __attribute__((ext_vector_type(16))) __bf16 v16bf;
typedef __attribute__((ext_vector_type(8)))  float  v8f;
typedef __attribute__((ext_vector_type(4)))  int    v4i;

struct FragBF { unsigned int u[8]; };

DEVI v8f wmma_bf(const FragBF& a, const FragBF& b, v8f c) {
  union { FragBF f; v16bf v; } A, Bv;
  A.f = a; Bv.f = b;
  return __builtin_amdgcn_wmma_f32_16x16x32_bf16(false, A.v, false, Bv.v,
                                                 (short)0, c, false, false);
}

DEVI unsigned short f2bf(float f) {
  unsigned int u = __float_as_uint(f);
  u += 0x7fffu + ((u >> 16) & 1u);          // round-to-nearest-even
  return (unsigned short)(u >> 16);
}

// 16B global -> LDS copy. CDNA5 path: async DMA straight to LDS (ASYNCcnt).
DEVI void cp16_lds(unsigned short* lds, const unsigned short* g) {
#if USE_ASYNC_LDS
  __builtin_amdgcn_global_load_async_to_lds_b128(
      (__attribute__((address_space(1))) v4i*)(g),
      (__attribute__((address_space(3))) v4i*)(lds), 0, 0);
#else
  *(uint4*)lds = *(const uint4*)g;
#endif
}
DEVI void async_fence_lds() {
#if USE_ASYNC_LDS
#if __has_builtin(__builtin_amdgcn_s_wait_asynccnt)
  __builtin_amdgcn_s_wait_asynccnt(0);
#else
  asm volatile("s_wait_asynccnt 0" ::: "memory");
#endif
#endif
}

// A fragment (16 x K32, bf16), ISA 7.12.2 layout:
// lane: row = lane&15; h=lane>>4; VGPR0-3 = K[8h..8h+7], VGPR4-7 = K[16+8h..23+8h]
DEVI void loadA(FragBF& f, const unsigned short* p, int ld, int lane) {
  const unsigned short* q = p + (size_t)(lane & 15) * ld + (lane >> 4) * 8;
  uint4 a = *(const uint4*)q;
  uint4 b = *(const uint4*)(q + 16);
  f.u[0]=a.x; f.u[1]=a.y; f.u[2]=a.z; f.u[3]=a.w;
  f.u[4]=b.x; f.u[5]=b.y; f.u[6]=b.z; f.u[7]=b.w;
}
// B fragment (K32 x 16) sourced from an (N,K)-row-major buffer:
// lane: n = lane&15 (row n of source); h=lane>>4; VGPR0-3 = K[8h..8h+7], VGPR4-7 = K[16+8h..23+8h]
DEVI void loadB(FragBF& f, const unsigned short* p, int ld, int lane) {
  const unsigned short* q = p + (size_t)(lane & 15) * ld + (lane >> 4) * 8;
  uint4 a = *(const uint4*)q;
  uint4 b = *(const uint4*)(q + 16);
  f.u[0]=a.x; f.u[1]=a.y; f.u[2]=a.z; f.u[3]=a.w;
  f.u[4]=b.x; f.u[5]=b.y; f.u[6]=b.z; f.u[7]=b.w;
}

// ---------------------------------------------------------------------------
// GEMM: out(M,N) = act( A(M,K)bf16 @ W(N,K)bf16^T + bias[n] + rowBias[m/rowDiv][n] )
// block tile 128x128, 8 waves (4x2), wave tile 32x64 -> 8 wmma per K-step of 32.
// Double-buffered async global->LDS tiles, one barrier per K-step.
// ---------------------------------------------------------------------------
__global__ __launch_bounds__(256) void gemm_bf16_kernel(
    const unsigned short* __restrict__ A, const unsigned short* __restrict__ W,
    const float* __restrict__ bias, const float* __restrict__ rowBias, int rowDiv,
    int M, int N, int K, int act,
    float* __restrict__ outF, unsigned short* __restrict__ outBF)
{
  __shared__ __align__(16) unsigned short As[2][128 * 40];
  __shared__ __align__(16) unsigned short Bs[2][128 * 40];
  const int tid  = threadIdx.x;
  const int lane = tid & 31, wv = tid >> 5;
  const int wm = wv & 3, wn = wv >> 2;          // 4x2 wave grid
  const int m0 = blockIdx.y * 128, n0 = blockIdx.x * 128;
  const int lm = lane & 15, lh = lane >> 4;

  auto issue = [&](int kt, int buf) {
#if USE_ASYNC_LDS
    #pragma unroll
    for (int it = 0; it < 2; ++it) {
      int idx = tid + it * 256;
      int row = idx >> 2, q = (idx & 3) * 8;
      int gr = m0 + row; gr = gr < M ? gr : M - 1;
      cp16_lds(&As[buf][row * 40 + q], A + (size_t)gr * K + kt * 32 + q);
      int gn = n0 + row; gn = gn < N ? gn : N - 1;
      cp16_lds(&Bs[buf][row * 40 + q], W + (size_t)gn * K + kt * 32 + q);
    }
#else
    uint4 ra[2], rb[2];
    #pragma unroll
    for (int it = 0; it < 2; ++it) {            // all loads first (ILP)
      int idx = tid + it * 256;
      int row = idx >> 2, q = (idx & 3) * 8;
      int gr = m0 + row; gr = gr < M ? gr : M - 1;
      ra[it] = *(const uint4*)(A + (size_t)gr * K + kt * 32 + q);
      int gn = n0 + row; gn = gn < N ? gn : N - 1;
      rb[it] = *(const uint4*)(W + (size_t)gn * K + kt * 32 + q);
    }
    #pragma unroll
    for (int it = 0; it < 2; ++it) {
      int idx = tid + it * 256;
      int row = idx >> 2, q = (idx & 3) * 8;
      *(uint4*)&As[buf][row * 40 + q] = ra[it];
      *(uint4*)&Bs[buf][row * 40 + q] = rb[it];
    }
#endif
  };

  v8f zero;
  #pragma unroll
  for (int e = 0; e < 8; ++e) zero[e] = 0.f;
  v8f acc[2][4];
  #pragma unroll
  for (int i = 0; i < 2; ++i)
    #pragma unroll
    for (int j = 0; j < 4; ++j) acc[i][j] = zero;

  const int nk = K >> 5;
  issue(0, 0);
  for (int t = 0; t < nk; ++t) {
    const int buf = t & 1;
    async_fence_lds();            // own async copies for tile t are in LDS
    __syncthreads();              // everyone's tile t visible; buf^1 free
    if (t + 1 < nk) issue(t + 1, buf ^ 1);
    FragBF fa[2], fb[4];
    #pragma unroll
    for (int i = 0; i < 2; ++i) loadA(fa[i], &As[buf][(wm * 32 + i * 16) * 40], 40, lane);
    #pragma unroll
    for (int j = 0; j < 4; ++j) loadB(fb[j], &Bs[buf][(wn * 64 + j * 16) * 40], 40, lane);
    #pragma unroll
    for (int i = 0; i < 2; ++i)
      #pragma unroll
      for (int j = 0; j < 4; ++j)
        acc[i][j] = wmma_bf(fa[i], fb[j], acc[i][j]);
    // no trailing barrier: buf is only overwritten at issue(t+2), which happens
    // after barrier(t+1); our LDS reads completed before the wmmas executed.
  }

  #pragma unroll
  for (int i = 0; i < 2; ++i)
    #pragma unroll
    for (int j = 0; j < 4; ++j) {
      int n = n0 + wn * 64 + j * 16 + lm;
      if (n >= N) continue;
      float bsum = bias ? bias[n] : 0.f;
      #pragma unroll
      for (int r = 0; r < 8; ++r) {
        int m = m0 + wm * 32 + i * 16 + r + 8 * lh;
        if (m >= M) continue;
        float v = acc[i][j][r] + bsum;
        if (rowBias) v += rowBias[(size_t)(m / rowDiv) * N + n];
        if (act == 1) v = v > 0.f ? v : __expf(v) - 1.f;   // ELU
        size_t off = (size_t)m * N + n;
        if (outBF) outBF[off] = f2bf(v);
        if (outF)  outF[off]  = v;
      }
    }
}

// ---------------------------------------------------------------------------
// Fused GLU -> +residual -> LayerNorm over H=256. One row per 256-thread block.
// resMap: 0 = residual row == row ; 1 = decoder row -> full (b*T + EL + t)
// ---------------------------------------------------------------------------
__global__ __launch_bounds__(256) void glu_res_ln_kernel(
    const float* __restrict__ Y, const float* __restrict__ Res, int resMap,
    const float* __restrict__ gam, const float* __restrict__ bet,
    float* __restrict__ outF, unsigned short* __restrict__ outBF,
    float* __restrict__ outDec)
{
  __shared__ float smA[8], smB[8], bcast[2];
  const int row = blockIdx.x, n = threadIdx.x;
  const int wv = threadIdx.x >> 5, l = threadIdx.x & 31;

  float a  = Y[(size_t)row * (2 * Hh) + n];
  float gt = Y[(size_t)row * (2 * Hh) + Hh + n];
  float v  = a / (1.f + __expf(-gt));            // a * sigmoid(gate)
  size_t resRow;
  if (resMap == 1) { int b = row / TDd, t = row % TDd; resRow = (size_t)b * Tt + ELl + t; }
  else resRow = (size_t)row;
  float x = v + Res[resRow * Hh + n];

  float s = x, s2 = x * x;
  for (int off = 16; off; off >>= 1) { s += __shfl_xor(s, off, 32); s2 += __shfl_xor(s2, off, 32); }
  if (l == 0) { smA[wv] = s; smB[wv] = s2; }
  __syncthreads();
  if (wv == 0) {
    float ts = (l < 8) ? smA[l] : 0.f, t2 = (l < 8) ? smB[l] : 0.f;
    for (int off = 4; off; off >>= 1) { ts += __shfl_xor(ts, off, 32); t2 += __shfl_xor(t2, off, 32); }
    if (l == 0) { bcast[0] = ts; bcast[1] = t2; }
  }
  __syncthreads();
  float mean = bcast[0] * (1.f / Hh);
  float var  = bcast[1] * (1.f / Hh) - mean * mean;
  float o = (x - mean) * rsqrtf(var + EPSc) * gam[n] + bet[n];

  if (outF)  outF[(size_t)row * Hh + n]  = o;
  if (outBF) outBF[(size_t)row * Hh + n] = f2bf(o);
  if (outDec) {
    int b = row / Tt, t = row % Tt;
    if (t >= ELl) outDec[((size_t)b * TDd + (t - ELl)) * Hh + n] = o;
  }
}

__global__ void cvt_kernel(const float* __restrict__ in, unsigned short* __restrict__ out, size_t n) {
  size_t i = (size_t)blockIdx.x * blockDim.x + threadIdx.x;
  size_t stride = (size_t)gridDim.x * blockDim.x;
  for (; i < n; i += stride) out[i] = f2bf(in[i]);
}

__global__ void mean_heads_kernel(const float* __restrict__ attn, unsigned short* __restrict__ out) {
  size_t i = (size_t)blockIdx.x * blockDim.x + threadIdx.x;
  size_t n = (size_t)RDd * DHh;
  if (i >= n) return;
  int d = (int)(i % DHh); size_t bt = i / DHh;
  int t = (int)(bt % TDd), b = (int)(bt / TDd);
  float s = 0.f;
  #pragma unroll
  for (int h = 0; h < NHh; ++h)
    s += attn[(((size_t)b * NHh + h) * TDd + t) * DHh + d];
  out[i] = f2bf(s * (1.f / NHh));
}

// ---------------------------------------------------------------------------
// Causal flash attention, decoder queries only (t >= EL).
// grid = (2 query-blocks of 128, NH, B); 8 waves x 16 query rows each.
// 32-key LDS tiles (K async-DMA'd, V transposed in-flight);
// S = Q@K^T (4 wmma), online softmax, P@V (4 wmma).
// ---------------------------------------------------------------------------
__global__ __launch_bounds__(256) void attn_kernel(
    const unsigned short* __restrict__ qkv,   // (R, 576) bf16: [q(256) k(256) v(64)]
    float* __restrict__ attnOut)              // (B, NH, TD, DH) fp32
{
  __shared__ __align__(16) unsigned short Ks[32 * 72];       // 32 keys x 64 dh
  __shared__ __align__(16) unsigned short Vt[64 * 40];       // 64 dh x 32 keys (transposed)
  __shared__ __align__(16) unsigned short Pw[8 * 16 * 40];   // per-wave P tile 16x32

  const int b = blockIdx.z, h = blockIdx.y, qb = blockIdx.x;
  const int tid = threadIdx.x, lane = tid & 31, wv = tid >> 5;
  const int qbase = ELl + qb * 128;
  const int qrow0 = qbase + wv * 16;
  const int lm = lane & 15, lh = lane >> 4;
  const float scale = 0.125f;                 // DH^-0.5

  FragBF fq[2];
  {
    const unsigned short* qp = qkv + (size_t)(b * Tt + qrow0) * 576 + h * DHh;
    loadA(fq[0], qp, 576, lane);
    loadA(fq[1], qp + 32, 576, lane);
  }

  v8f zero;
  #pragma unroll
  for (int e = 0; e < 8; ++e) zero[e] = 0.f;
  v8f o[4];
  #pragma unroll
  for (int f = 0; f < 4; ++f) o[f] = zero;
  float mrun[8], lrun[8];
  #pragma unroll
  for (int r = 0; r < 8; ++r) { mrun[r] = -1e30f; lrun[r] = 0.f; }

  const int nIter = (qbase + 128) / 32;
  for (int it = 0; it < nIter; ++it) {
    const int s0 = it * 32;
    {   // K tile via async DMA to LDS; V transposed through VGPRs
      int i = tid >> 3, q = (tid & 7) * 8;
      const unsigned short* base = qkv + (size_t)(b * Tt + s0 + i) * 576;
      cp16_lds(&Ks[i * 72 + q], base + NHh * DHh + h * DHh + q);
      uint4 vv = *(const uint4*)(base + 2 * NHh * DHh + q);
      unsigned int uu[4] = { vv.x, vv.y, vv.z, vv.w };
      #pragma unroll
      for (int e = 0; e < 4; ++e) {
        Vt[(q + 2 * e    ) * 40 + i] = (unsigned short)(uu[e] & 0xffffu);
        Vt[(q + 2 * e + 1) * 40 + i] = (unsigned short)(uu[e] >> 16);
      }
    }
    async_fence_lds();
    __syncthreads();
    if (s0 <= qrow0 + 15) {         // wave-uniform causal skip
      v8f s[2]; s[0] = zero; s[1] = zero;
      #pragma unroll
      for (int j = 0; j < 2; ++j) {
        FragBF fb0, fb1;
        loadB(fb0, &Ks[(j * 16) * 72 + 0], 72, lane);
        s[j] = wmma_bf(fq[0], fb0, s[j]);
        loadB(fb1, &Ks[(j * 16) * 72 + 32], 72, lane);
        s[j] = wmma_bf(fq[1], fb1, s[j]);
      }
      float pv0[8], pv1[8], alpha[8];
      #pragma unroll
      for (int r = 0; r < 8; ++r) {
        int t = qrow0 + r + 8 * lh;
        float v0 = s[0][r] * scale; if (s0 + lm      > t) v0 = -1e30f;
        float v1 = s[1][r] * scale; if (s0 + 16 + lm > t) v1 = -1e30f;
        float rm = fmaxf(v0, v1);
        for (int off = 1; off < 16; off <<= 1) rm = fmaxf(rm, __shfl_xor(rm, off, 32));
        float mn = fmaxf(mrun[r], rm);
        float al = __expf(mrun[r] - mn);
        float p0 = __expf(v0 - mn), p1 = __expf(v1 - mn);
        float rs = p0 + p1;
        for (int off = 1; off < 16; off <<= 1) rs += __shfl_xor(rs, off, 32);
        lrun[r] = lrun[r] * al + rs;
        mrun[r] = mn; alpha[r] = al; pv0[r] = p0; pv1[r] = p1;
      }
      #pragma unroll
      for (int f = 0; f < 4; ++f)
        #pragma unroll
        for (int r = 0; r < 8; ++r) o[f][r] *= alpha[r];
      unsigned short* pp = &Pw[wv * 16 * 40];     // C-layout -> A-layout via LDS
      #pragma unroll
      for (int r = 0; r < 8; ++r) {
        int rr = r + 8 * lh;
        pp[rr * 40 + lm]      = f2bf(pv0[r]);
        pp[rr * 40 + 16 + lm] = f2bf(pv1[r]);
      }
      FragBF fp;
      loadA(fp, pp, 40, lane);
      #pragma unroll
      for (int f = 0; f < 4; ++f) {
        FragBF fv;
        loadB(fv, &Vt[(f * 16) * 40], 40, lane);
        o[f] = wmma_bf(fp, fv, o[f]);
      }
    }
    __syncthreads();
  }
  #pragma unroll
  for (int f = 0; f < 4; ++f)
    #pragma unroll
    for (int r = 0; r < 8; ++r) {
      int m  = r + 8 * lh;
      int tq = qb * 128 + wv * 16 + m;
      int d  = f * 16 + lm;
      attnOut[(((size_t)b * NHh + h) * TDd + tq) * DHh + d] = o[f][r] / lrun[r];
    }
}

} // namespace

extern "C" void kernel_launch(void* const* d_in, const int* in_sizes, int n_in,
                              void* d_out, int out_size, void* d_ws, size_t ws_size,
                              hipStream_t stream) {
  (void)in_sizes; (void)n_in; (void)out_size; (void)ws_size;
  const float* tf       = (const float*)d_in[0];
  const float* ce       = (const float*)d_in[1];
  const float* eg_Wa    = (const float*)d_in[2];
  const float* eg_ba    = (const float*)d_in[3];
  const float* eg_Wc    = (const float*)d_in[4];
  const float* eg_Wi    = (const float*)d_in[5];
  const float* eg_bi    = (const float*)d_in[6];
  const float* eg_Wg    = (const float*)d_in[7];
  const float* eg_bg    = (const float*)d_in[8];
  const float* eg_lng   = (const float*)d_in[9];
  const float* eg_lnb   = (const float*)d_in[10];
  const float* att_Wqkv = (const float*)d_in[11];
  const float* att_Wout = (const float*)d_in[12];
  const float* ag_W     = (const float*)d_in[13];
  const float* ag_b     = (const float*)d_in[14];
  const float* aln_g    = (const float*)d_in[15];
  const float* aln_b    = (const float*)d_in[16];
  const float* pg_Wa    = (const float*)d_in[17];
  const float* pg_ba    = (const float*)d_in[18];
  const float* pg_Wi    = (const float*)d_in[19];
  const float* pg_bi    = (const float*)d_in[20];
  const float* pg_Wg    = (const float*)d_in[21];
  const float* pg_bg    = (const float*)d_in[22];
  const float* pg_lng   = (const float*)d_in[23];
  const float* pg_lnb   = (const float*)d_in[24];
  const float* dg_W     = (const float*)d_in[25];
  const float* dg_b     = (const float*)d_in[26];
  const float* dln_g    = (const float*)d_in[27];
  const float* dln_b    = (const float*)d_in[28];

  // ---- workspace arena (~380 MB, 256B aligned slots) ----
  char* ws = (char*)d_ws;
  size_t off = 0;
  auto take = [&](size_t bytes) -> char* {
    char* p = ws + off;
    off = (off + bytes + 255) & ~(size_t)255;
    return p;
  };
  unsigned short* wWa  = (unsigned short*)take((size_t)Hh * Hh * 2);
  unsigned short* wWc  = (unsigned short*)take((size_t)Hh * Hh * 2);
  unsigned short* wWi  = (unsigned short*)take((size_t)Hh * Hh * 2);
  unsigned short* wWg  = (unsigned short*)take((size_t)2 * Hh * Hh * 2);
  unsigned short* wQkv = (unsigned short*)take((size_t)576 * Hh * 2);
  unsigned short* wOut = (unsigned short*)take((size_t)Hh * DHh * 2);
  unsigned short* wAg  = (unsigned short*)take((size_t)2 * Hh * Hh * 2);
  unsigned short* wPa  = (unsigned short*)take((size_t)Hh * Hh * 2);
  unsigned short* wPi  = (unsigned short*)take((size_t)Hh * Hh * 2);
  unsigned short* wPg  = (unsigned short*)take((size_t)2 * Hh * Hh * 2);
  unsigned short* wDg  = (unsigned short*)take((size_t)2 * Hh * Hh * 2);
  unsigned short* ceBf = (unsigned short*)take((size_t)Bq * Hh * 2);
  float* ceProj = (float*)take((size_t)Bq * Hh * 4);
  float* big    = (float*)take((size_t)Rr * 2 * Hh * 4);            // reused GLU input
  unsigned short* bfA = (unsigned short*)take((size_t)Rr * Hh * 2); // tf_bf -> enr_bf
  unsigned short* bfB = (unsigned short*)take((size_t)Rr * Hh * 2); // xa -> dec slots
  unsigned short* bfC = (unsigned short*)take((size_t)Rr * Hh * 2); // xi -> dec slots
  unsigned short* qkvBf = (unsigned short*)take((size_t)Rr * 576 * 2);
  float* attnF  = (float*)take((size_t)Bq * NHh * TDd * DHh * 4);
  float* enrDec = (float*)take((size_t)RDd * Hh * 4);
  float* xF     = (float*)take((size_t)RDd * Hh * 4);
  float* x2F    = (float*)take((size_t)RDd * Hh * 4);
  unsigned short* aBf  = bfB;
  unsigned short* xBf  = bfB + (size_t)RDd * Hh;
  unsigned short* y1Bf = bfB + (size_t)2 * RDd * Hh;
  unsigned short* y2Bf = bfB + (size_t)3 * RDd * Hh;
  unsigned short* mattnBf = bfC;
  unsigned short* x2Bf    = bfC + (size_t)RDd * Hh;

  auto cvt = [&](const float* src, unsigned short* dst, size_t n) {
    int blocks = (int)((n + 255) / 256); if (blocks > 4096) blocks = 4096;
    cvt_kernel<<<blocks, 256, 0, stream>>>(src, dst, n);
  };
  auto gemm = [&](const unsigned short* A, const unsigned short* W, const float* bias,
                  const float* rowBias, int rowDiv, int M, int N, int K, int act,
                  float* outF, unsigned short* outBF) {
    dim3 grid((N + 127) / 128, (M + 127) / 128);
    gemm_bf16_kernel<<<grid, 256, 0, stream>>>(A, W, bias, rowBias, rowDiv,
                                               M, N, K, act, outF, outBF);
  };

  // ---- bf16 conversions ----
  cvt(tf, bfA, (size_t)Rr * Hh);
  cvt(ce, ceBf, (size_t)Bq * Hh);
  cvt(eg_Wa, wWa, (size_t)Hh * Hh);   cvt(eg_Wc, wWc, (size_t)Hh * Hh);
  cvt(eg_Wi, wWi, (size_t)Hh * Hh);   cvt(eg_Wg, wWg, (size_t)2 * Hh * Hh);
  cvt(att_Wqkv, wQkv, (size_t)576 * Hh); cvt(att_Wout, wOut, (size_t)Hh * DHh);
  cvt(ag_W, wAg, (size_t)2 * Hh * Hh);
  cvt(pg_Wa, wPa, (size_t)Hh * Hh);   cvt(pg_Wi, wPi, (size_t)Hh * Hh);
  cvt(pg_Wg, wPg, (size_t)2 * Hh * Hh); cvt(dg_W, wDg, (size_t)2 * Hh * Hh);

  // ---- encoder gating ----
  gemm(ceBf, wWc, nullptr, nullptr, 1, Bq, Hh, Hh, 0, ceProj, nullptr);            // ce@Wc^T
  gemm(bfA, wWa, eg_ba, ceProj, Tt, Rr, Hh, Hh, 1, nullptr, bfB);                  // elu(tf@Wa^T+ba+ceProj[b])
  gemm(bfB, wWi, eg_bi, nullptr, 1, Rr, Hh, Hh, 0, nullptr, bfC);                  // @Wi^T+bi
  gemm(bfC, wWg, eg_bg, nullptr, 1, Rr, 2 * Hh, Hh, 0, big, nullptr);              // @Wg^T+bg
  glu_res_ln_kernel<<<Rr, 256, 0, stream>>>(big, tf, 0, eg_lng, eg_lnb,
                                            nullptr, bfA, enrDec);                 // enriched

  // ---- attention (decoder queries only) ----
  gemm(bfA, wQkv, nullptr, nullptr, 1, Rr, 576, Hh, 0, nullptr, qkvBf);            // qkv
  attn_kernel<<<dim3(2, NHh, Bq), 256, 0, stream>>>(qkvBf, attnF);
  { size_t n = (size_t)RDd * DHh;
    mean_heads_kernel<<<(unsigned)((n + 255) / 256), 256, 0, stream>>>(attnF, mattnBf); }
  gemm(mattnBf, wOut, nullptr, nullptr, 1, RDd, Hh, DHh, 0, nullptr, aBf);         // m_attn@Wout^T

  // ---- decoder ----
  gemm(aBf, wAg, ag_b, nullptr, 1, RDd, 2 * Hh, Hh, 0, big, nullptr);
  glu_res_ln_kernel<<<RDd, 256, 0, stream>>>(big, enrDec, 0, aln_g, aln_b,
                                             xF, xBf, nullptr);
  gemm(xBf, wPa, pg_ba, nullptr, 1, RDd, Hh, Hh, 1, nullptr, y1Bf);                // elu
  gemm(y1Bf, wPi, pg_bi, nullptr, 1, RDd, Hh, Hh, 0, nullptr, y2Bf);
  gemm(y2Bf, wPg, pg_bg, nullptr, 1, RDd, 2 * Hh, Hh, 0, big, nullptr);
  glu_res_ln_kernel<<<RDd, 256, 0, stream>>>(big, xF, 0, pg_lng, pg_lnb,
                                             x2F, x2Bf, nullptr);
  gemm(x2Bf, wDg, dg_b, nullptr, 1, RDd, 2 * Hh, Hh, 0, big, nullptr);
  glu_res_ln_kernel<<<RDd, 256, 0, stream>>>(big, tf, 1, dln_g, dln_b,
                                             (float*)d_out, nullptr, nullptr);
}
